// CorrespondenceGenerationArch_9371618640389
// MI455X (gfx1250) — matrix operand order repaired and checked by reading, not compiled, not run on hardware
//
#include <hip/hip_runtime.h>
#include <hip/hip_bf16.h>

// ---------------- problem constants (B=1 image; batched via host loop) ------
#define FH   72
#define FW   72
#define FC   256
#define HO   70
#define WO   70
#define NPOS (HO * WO)            // 4900 patches per image
#define POS  (FH * FW)            // 5184 spatial positions
#define COLT ((NPOS + 15) / 16)   // 307 column tiles
#define ROWT ((NPOS + 31) / 32)   // 154 workgroups (32 rows each)
#define OH   (4 * FH)             // 288
#define OW   (4 * FW)             // 288

typedef __attribute__((ext_vector_type(16))) _Float16 v16h;
typedef __attribute__((ext_vector_type(8)))  _Float16 v8h;
typedef __attribute__((ext_vector_type(8)))  float    v8f;

union F16V { v16h v; v8h h[2]; };

#define WMMA_F16(A, B, Cc) \
  __builtin_amdgcn_wmma_f32_16x16x32_f16(false, (A), false, (B), (short)0, (Cc), false, false)

// ---------------------------------------------------------------------------
// Kernel 1: channel-L2-normalize (x16 scale) and split f32 -> (hi,lo) f16,
// storing position-major / channel-contiguous layout [y*FW+x][c].
// grid = (POS, 2): blockIdx.y selects feature map 1 or 2. 256 threads = FC.
// ---------------------------------------------------------------------------
__global__ __launch_bounds__(256) void prep_normalize_split(
    const float* __restrict__ f1, const float* __restrict__ f2,
    _Float16* __restrict__ n1hi, _Float16* __restrict__ n1lo,
    _Float16* __restrict__ n2hi, _Float16* __restrict__ n2lo) {
  const int pos = blockIdx.x;     // 0..POS-1
  const int c   = threadIdx.x;    // 0..FC-1
  const float* f = blockIdx.y ? f2 : f1;
  _Float16* oh   = blockIdx.y ? n2hi : n1hi;
  _Float16* ol   = blockIdx.y ? n2lo : n1lo;

  const float v = f[c * POS + pos];   // input layout [c][y][x]

  __shared__ float red[256];
  red[c] = v * v;
  __syncthreads();
  for (int s = 128; s > 0; s >>= 1) {
    if (c < s) red[c] += red[c + s];
    __syncthreads();
  }
  const float nrm = sqrtf(red[0]);
  const float inv = 16.0f / fmaxf(nrm, 1e-12f);   // x16: keep lo-part in f16 normal range
  const float sv  = v * inv;
  const _Float16 hi = (_Float16)sv;
  const float    lo = sv - (float)hi;             // exact in f32
  oh[pos * FC + c] = hi;
  ol[pos * FC + c] = (_Float16)lo;
}

// ---------------------------------------------------------------------------
// Kernel 2: fused patch cross-correlation GEMM (implicit im2col) + row argmax.
// One workgroup = 8 wave32 = 32 output rows (2 row-tiles per wave).
// Each wave sweeps column tiles ct = wave, wave+8, ... keeping a running
// per-lane max; LDS merge at the end with first-occurrence tie-break.
// Error-compensated f16 split: acc += Ahi*Bhi + Ahi*Blo + Alo*Bhi.
// ---------------------------------------------------------------------------
__global__ __launch_bounds__(256) void corr_argmax(
    const _Float16* __restrict__ Ahi, const _Float16* __restrict__ Alo,
    const _Float16* __restrict__ Bhi, const _Float16* __restrict__ Blo,
    int* __restrict__ out_idx) {
  __shared__ float s_val[8][32][16];
  __shared__ int   s_idx[8][32][16];

  const int tid  = threadIdx.x;
  const int wave = tid >> 5;     // 0..7
  const int lane = tid & 31;
  const int half = lane >> 4;    // 0/1: which 16-lane half
  const int lm   = lane & 15;

  const int rowbase = blockIdx.x * 32;

  int r0 = rowbase + lm;       if (r0 > NPOS - 1) r0 = NPOS - 1;
  int r1 = rowbase + 16 + lm;  if (r1 > NPOS - 1) r1 = NPOS - 1;
  const int a0Base = ((r0 / WO) * FW + (r0 % WO)) * FC;
  const int a1Base = ((r1 / WO) * FW + (r1 % WO)) * FC;

  float best0v[8], best1v[8];
  int   best0i[8], best1i[8];
#pragma unroll
  for (int v = 0; v < 8; ++v) {
    best0v[v] = -3.0e38f; best1v[v] = -3.0e38f;
    best0i[v] = 0;        best1i[v] = 0;
  }

  for (int ct = wave; ct < COLT; ct += 8) {
    int j  = ct * 16 + lm;
    int jc = (j > NPOS - 1) ? (NPOS - 1) : j;            // clamp pad cols (exact dup)
    const int bBase = ((jc / WO) * FW + (jc % WO)) * FC;

    v8f acc0 = {};
    v8f acc1 = {};

#pragma unroll 1
    for (int dydx = 0; dydx < 9; ++dydx) {               // K outer: 3x3 patch offset
      const int off = ((dydx / 3) * FW + (dydx % 3)) * FC;
      // Per-lane fragment base pointers; layouts match ISA 7.12.2:
      //  A 16x32 f16: lane<16 -> K{0..7,16..23}, lane>=16 -> K{8..15,24..31}
      //  B 32x16 f16: lane<16 -> K{0..7,16..23}, lane>=16 -> K{8..15,24..31}
      const _Float16* pA0h = Ahi + a0Base + off + 8 * half;
      const _Float16* pA0l = Alo + a0Base + off + 8 * half;
      const _Float16* pA1h = Ahi + a1Base + off + 8 * half;
      const _Float16* pA1l = Alo + a1Base + off + 8 * half;
      const _Float16* pBh  = Bhi + bBase + off + 8 * half;
      const _Float16* pBl  = Blo + bBase + off + 8 * half;

#pragma unroll 2
      for (int c0 = 0; c0 < FC; c0 += 32) {              // K inner: 32 channels
        F16V a0h, a0l, a1h, a1l, bh, bl;
        a0h.h[0] = *(const v8h*)(pA0h + c0);
        a0h.h[1] = *(const v8h*)(pA0h + c0 + 16);
        a0l.h[0] = *(const v8h*)(pA0l + c0);
        a0l.h[1] = *(const v8h*)(pA0l + c0 + 16);
        a1h.h[0] = *(const v8h*)(pA1h + c0);
        a1h.h[1] = *(const v8h*)(pA1h + c0 + 16);
        a1l.h[0] = *(const v8h*)(pA1l + c0);
        a1l.h[1] = *(const v8h*)(pA1l + c0 + 16);
        bh.h[0]  = *(const v8h*)(pBh + c0);
        bh.h[1]  = *(const v8h*)(pBh + c0 + 16);
        bl.h[0]  = *(const v8h*)(pBl + c0);
        bl.h[1]  = *(const v8h*)(pBl + c0 + 16);

        acc0 = WMMA_F16(a0h.v, bh.v, acc0);
        acc0 = WMMA_F16(a0h.v, bl.v, acc0);
        acc0 = WMMA_F16(a0l.v, bh.v, acc0);
        acc1 = WMMA_F16(a1h.v, bh.v, acc1);
        acc1 = WMMA_F16(a1h.v, bl.v, acc1);
        acc1 = WMMA_F16(a1l.v, bh.v, acc1);
      }
    }

    // C layout: VGPR v, lane -> row (v + 8*half), col N = lm (global col jc).
    // Strict '>' keeps the smallest column index within this lane's residue class.
#pragma unroll
    for (int v = 0; v < 8; ++v) {
      if (acc0[v] > best0v[v]) { best0v[v] = acc0[v]; best0i[v] = jc; }
      if (acc1[v] > best1v[v]) { best1v[v] = acc1[v]; best1i[v] = jc; }
    }
  }

#pragma unroll
  for (int v = 0; v < 8; ++v) {
    s_val[wave][v + 8 * half][lm]      = best0v[v];
    s_idx[wave][v + 8 * half][lm]      = best0i[v];
    s_val[wave][16 + v + 8 * half][lm] = best1v[v];
    s_idx[wave][16 + v + 8 * half][lm] = best1i[v];
  }
  __syncthreads();

  if (tid < 32) {                     // one thread per output row of this WG
    float bv = -3.0e38f;
    int   bi = 0;
    for (int wv = 0; wv < 8; ++wv) {
#pragma unroll 1
      for (int n = 0; n < 16; ++n) {
        const float vv = s_val[wv][tid][n];
        const int   ii = s_idx[wv][tid][n];
        if (vv > bv || (vv == bv && ii < bi)) { bv = vv; bi = ii; }
      }
    }
    const int R = rowbase + tid;
    if (R < NPOS) out_idx[R] = bi;
  }
}

// ---------------------------------------------------------------------------
// Kernel 3: index -> flow, pad to 72x72, 4x nearest upsample * 4, 9 shifts.
// One thread per (k, h, w) output pixel; writes both flow channels.
// ---------------------------------------------------------------------------
__global__ __launch_bounds__(256) void build_flow(
    const int* __restrict__ idx, float* __restrict__ out) {
  const int t = blockIdx.x * blockDim.x + threadIdx.x;   // over 9*OH*OW
  if (t >= 9 * OH * OW) return;
  const int w = t % OW;
  const int h = (t / OW) % OH;
  const int k = t / (OH * OW);
  const int si = k / 3, sj = k % 3;                      // shift (4*si, 4*sj)

  float fw = 0.0f, fh = 0.0f;
  const int hs = h - 4 * si, ws = w - 4 * sj;
  if (hs >= 0 && ws >= 0) {
    const int hh = hs >> 2, ww = ws >> 2;                // 0..71
    if (hh < HO && ww < WO) {                            // zero-pad region otherwise
      const int m = idx[hh * WO + ww];
      fw = 4.0f * (float)(m % WO - ww);
      fh = 4.0f * (float)(m / WO - hh);
    }
  }
  float* o = out + (size_t)t * 2;
  o[0] = fw;
  o[1] = fh;
}

// ---------------------------------------------------------------------------
extern "C" void kernel_launch(void* const* d_in, const int* in_sizes, int n_in,
                              void* d_out, int out_size, void* d_ws, size_t ws_size,
                              hipStream_t stream) {
  const float* f1 = (const float*)d_in[0];   // (B, 256, 72, 72)
  const float* f2 = (const float*)d_in[1];   // (B, 256, 72, 72)
  (void)d_in; (void)n_in; (void)ws_size;

  const int imgElems = FC * FH * FW;                 // per-image feature elements
  const int B = in_sizes[0] / imgElems;

  char* ws = (char*)d_ws;
  const size_t featBytes = (size_t)POS * FC * sizeof(_Float16);  // 2,654,208 (256B-aligned)
  _Float16* n1hi = (_Float16*)(ws);
  _Float16* n1lo = (_Float16*)(ws + 1 * featBytes);
  _Float16* n2hi = (_Float16*)(ws + 2 * featBytes);
  _Float16* n2lo = (_Float16*)(ws + 3 * featBytes);
  int*      midx = (int*)     (ws + 4 * featBytes);

  for (int b = 0; b < B; ++b) {
    const float* f1b = f1 + (size_t)b * imgElems;
    const float* f2b = f2 + (size_t)b * imgElems;
    float* outb = (float*)d_out + (size_t)b * 9 * OH * OW * 2;

    dim3 gPrep(POS, 2);
    prep_normalize_split<<<gPrep, 256, 0, stream>>>(f1b, f2b, n1hi, n1lo, n2hi, n2lo);

    corr_argmax<<<ROWT, 256, 0, stream>>>(n1hi, n1lo, n2hi, n2lo, midx);

    const int total = 9 * OH * OW;
    build_flow<<<(total + 255) / 256, 256, 0, stream>>>(midx, outb);
  }
}
